// ANFIS_85203561218832
// MI455X (gfx1250) — matrix operand order repaired and checked by reading, not compiled, not run on hardware
//
#include <hip/hip_runtime.h>

typedef _Float16 f16;
typedef __attribute__((ext_vector_type(16))) _Float16 v16h;
typedef __attribute__((ext_vector_type(8)))  float    v8f;

// Problem constants (from reference)
#define MFS   512                // membership functions per feature
#define BATCH 256
#define NCAT  1536               // 3 * 512 concatenated GEMM columns (W0,W1,Wb)

// Workspace layout (bytes)
#define OFF_AMEM   0                           // feature-0 memberships, f16 [256][512] row-major
#define OFF_CSWZ   (256*512*2)                 // feature-1 memberships, WMMA-A-fragment swizzled
#define OFF_BSWZ   (2*256*512*2)               // packed weights, WMMA-B-fragment swizzled: 512*1536 f16
#define OFF_SA     (OFF_BSWZ + 512*1536*2)     // 256 f32
#define OFF_SC     (OFF_SA + 1024)             // 256 f32
#define OFF_NUMER  (OFF_SC + 1024)             // 3*256 f32 bilinear-form accumulators
// total = 2,102,272 bytes (~2.0 MiB)

// ---------------------------------------------------------------------------
// Kernel 1: Gaussian membership layer + row sums + accumulator zeroing.
// Feature 0 -> plain row-major f16 (read by GEMM epilogue as A[b,i]).
// Feature 1 -> stored directly in WMMA A-fragment layout (16x32 f16 tile):
//   lane = ((K>>3)&1)*16 + (M&15),  e = (K>>4)*8 + (K&7)   [ISA 7.12.2]
// ---------------------------------------------------------------------------
__global__ void anfis_membership(const float* __restrict__ x,
                                 const float* __restrict__ mean,
                                 const float* __restrict__ sigma,
                                 f16* __restrict__ Amem,
                                 f16* __restrict__ Cswz,
                                 float* __restrict__ Sa,
                                 float* __restrict__ Sc,
                                 float* __restrict__ numer) {
    __shared__ float red[256];
    const int blk = blockIdx.x;           // 0..511
    const int f   = blk >> 8;             // feature
    const int b   = blk & 255;            // batch row
    const int tid = threadIdx.x;

    if (blk == 0) {                       // zero bilinear accumulators each launch
        numer[tid]       = 0.0f;
        numer[256 + tid] = 0.0f;
        numer[512 + tid] = 0.0f;
    }

    const float xv = x[b * 2 + f];
    float psum = 0.0f;
#pragma unroll
    for (int s = 0; s < 2; ++s) {
        const int m  = tid + s * 256;
        const float t  = (xv - mean[f * MFS + m]) / sigma[f * MFS + m];
        const float val = __expf(-t * t);
        psum += val;
        if (f == 0) {
            Amem[b * MFS + m] = (f16)val;
        } else {
            const int bt     = b >> 4;
            const int jt     = m >> 5;
            const int jj     = m & 31;
            const int laneHi = (jj >> 3) & 1;
            const int e      = ((jj >> 4) << 3) | (jj & 7);
            const int lane   = (laneHi << 4) | (b & 15);
            Cswz[(((bt * 16 + jt) * 32) + lane) * 16 + e] = (f16)val;
        }
    }

    red[tid] = psum;
    __syncthreads();
    for (int s = 128; s > 0; s >>= 1) {
        if (tid < s) red[tid] += red[tid + s];
        __syncthreads();
    }
    if (tid == 0) {
        if (f == 0) Sa[b] = red[0]; else Sc[b] = red[0];
    }
}

// ---------------------------------------------------------------------------
// Kernel 2: pack cw/cb into WMMA B-fragment layout (16-bit B 32x16 tile:
//   lane = laneHi*16 + N,  K = (e>>3)*16 + laneHi*8 + (e&7)), fragment-major so each lane's
// 16 halves are one contiguous 32-byte chunk -> fully coalesced v16h loads.
//   column n = k*512 + i, row K = j, value W_k[i,j] = cw[(i*512+j)*2+k] / cb.
// ---------------------------------------------------------------------------
__global__ void anfis_packB(const float* __restrict__ cw,
                            const float* __restrict__ cb,
                            f16* __restrict__ Bswz) {
    const int tid  = blockIdx.x * 256 + threadIdx.x;   // < 512*1536 = 786432
    const int e    = tid & 15;
    const int lane = (tid >> 4) & 31;
    const int tile = tid >> 9;              // 0..1535  (= nt*16 + jt)
    const int jt   = tile & 15;
    const int nt   = tile >> 4;             // 0..95
    const int n    = nt * 16 + (lane & 15);
    const int j    = jt * 32 + ((e >> 3) << 4) + ((lane >> 4) << 3) + (e & 7);
    const int k    = n >> 9;                // 0,1 -> cw column; 2 -> cb
    const int i    = n & 511;
    const float v  = (k < 2) ? cw[(i * MFS + j) * 2 + k] : cb[i * MFS + j];
    Bswz[tid] = (f16)v;
}

// ---------------------------------------------------------------------------
// Kernel 3: WMMA GEMM  D[b, n] = sum_j C[b,j] * Bcat[j,n], K split in halves
// across blocks (epilogue accumulation is linear in partial sums), fused with
// the epilogue  numer[k][b] += sum_i A[b,i] * D[b, k*512+i].
//
// Block = 128 threads (4 waves, wave32). All 4 waves share one batch tile bt
// and one K-half: the 8 shared A fragments (8 KB) are staged into LDS with
// CDNA5 async global->LDS copies (ASYNCcnt), then the inner loop reads A via
// ds_load and double-buffers the per-wave B fragments in registers so global
// load latency overlaps each 4-WMMA group.
// ---------------------------------------------------------------------------
__global__ void anfis_gemm(const f16*  __restrict__ Amem,
                           const f16*  __restrict__ CswzRaw,
                           const v16h* __restrict__ Bswz,
                           float* __restrict__ numer) {
    __shared__ __align__(32) f16 Alds[8 * 32 * 16];   // 8 K-tiles * 1KB

    const int lane = threadIdx.x & 31;
    const int wave = threadIdx.x >> 5;
    const int bt   = blockIdx.x / 12;                // 0..15 batch tile
    const int rem  = blockIdx.x % 12;
    const int kh   = rem / 6;                        // 0..1  K half
    const int ng   = (rem % 6) * 4 + wave;           // 0..23 n-group
    const int nt0  = ng * 4;
    const int jt0  = kh * 8;                         // first K tile

    // ---- Stage 8 A-fragment tiles (8192 B) into LDS: 128 threads x 4 x b128
    {
        const unsigned long long gsrc =
            (unsigned long long)(CswzRaw + (size_t)(bt * 16 + jt0) * 512); // 512 f16 per tile
        const unsigned ldsbase = (unsigned)(unsigned long long)(&Alds[0]);
#pragma unroll
        for (int t = 0; t < 4; ++t) {
            const unsigned       loff = ldsbase + (unsigned)((threadIdx.x + t * 128) * 16);
            const unsigned long long g = gsrc + (unsigned long long)((threadIdx.x + t * 128) * 16);
            asm volatile("global_load_async_to_lds_b128 %0, %1, off"
                         :: "v"(loff), "v"(g) : "memory");
        }
        asm volatile("s_wait_asynccnt 0" ::: "memory");
        __syncthreads();
    }
    const v16h* Afrag = (const v16h*)Alds;           // [8 tiles][32 lanes]

    v8f acc[4] = {v8f{}, v8f{}, v8f{}, v8f{}};

    // ---- prologue: fragments for first K step
    v16h acur = Afrag[lane];
    v16h bcur[4], bnxt[4];
#pragma unroll
    for (int q = 0; q < 4; ++q)
        bcur[q] = Bswz[((nt0 + q) * 16 + jt0) * 32 + lane];

    v16h anxt = acur;
#pragma unroll
    for (int jl = 0; jl < 8; ++jl) {
        if (jl < 7) {                                // fetch next K step first
            anxt = Afrag[(jl + 1) * 32 + lane];
#pragma unroll
            for (int q = 0; q < 4; ++q)
                bnxt[q] = Bswz[((nt0 + q) * 16 + jt0 + jl + 1) * 32 + lane];
        }
#pragma unroll
        for (int q = 0; q < 4; ++q)
            acc[q] = __builtin_amdgcn_wmma_f32_16x16x32_f16(
                false, acur, false, bcur[q], (short)0, acc[q], false, false);
        acur = anxt;
#pragma unroll
        for (int q = 0; q < 4; ++q)
            bcur[q] = bnxt[q];
    }

    // ---- Epilogue: weight by A[b,i], reduce over the 16 columns (lanes of
    // each half hold N=0..15 for rows M = laneHi*8 + v), accumulate.
    const int col     = lane & 15;
    const int laneHi  = lane >> 4;
    const int rowbase = bt * 16 + laneHi * 8;

#pragma unroll
    for (int q = 0; q < 4; ++q) {
        const int nt = nt0 + q;
        const int k  = (nt * 16) >> 9;       // which bilinear form (0,1,2)
        const int i0 = (nt * 16) & 511;
        float s[8];
#pragma unroll
        for (int v = 0; v < 8; ++v) {
            const float av = (float)Amem[(rowbase + v) * MFS + i0 + col];
            s[v] = acc[q][v] * av;
        }
        // butterfly reduction within each 16-lane half (wave32)
#pragma unroll
        for (int off = 1; off < 16; off <<= 1) {
#pragma unroll
            for (int v = 0; v < 8; ++v)
                s[v] += __shfl_xor(s[v], off, 32);
        }
        if (col == 0) {
#pragma unroll
            for (int v = 0; v < 8; ++v)
                atomicAdd(&numer[k * 256 + rowbase + v], s[v]);
        }
    }
}

// ---------------------------------------------------------------------------
// Kernel 4: out[b] = (x0*n0 + x1*n1 + n2) / (Sa*Sc)
// ---------------------------------------------------------------------------
__global__ void anfis_final(const float* __restrict__ x,
                            const float* __restrict__ Sa,
                            const float* __restrict__ Sc,
                            const float* __restrict__ numer,
                            float* __restrict__ out) {
    const int b = threadIdx.x;
    const float num = x[2 * b] * numer[b]
                    + x[2 * b + 1] * numer[256 + b]
                    + numer[512 + b];
    out[b] = num / (Sa[b] * Sc[b]);
}

extern "C" void kernel_launch(void* const* d_in, const int* in_sizes, int n_in,
                              void* d_out, int out_size, void* d_ws, size_t ws_size,
                              hipStream_t stream) {
    const float* x     = (const float*)d_in[0];   // [256,2]
    const float* mean  = (const float*)d_in[1];   // [2,512]
    const float* sigma = (const float*)d_in[2];   // [2,512]
    const float* cw    = (const float*)d_in[3];   // [262144,2]
    const float* cb    = (const float*)d_in[4];   // [262144,1]
    float* out = (float*)d_out;                   // [256,1]

    char* ws = (char*)d_ws;
    f16*   Amem  = (f16*)(ws + OFF_AMEM);
    f16*   Cswz  = (f16*)(ws + OFF_CSWZ);
    f16*   Bswz  = (f16*)(ws + OFF_BSWZ);
    float* Sa    = (float*)(ws + OFF_SA);
    float* Sc    = (float*)(ws + OFF_SC);
    float* numer = (float*)(ws + OFF_NUMER);

    anfis_membership<<<512, 256, 0, stream>>>(x, mean, sigma, Amem, Cswz, Sa, Sc, numer);
    anfis_packB<<<3072, 256, 0, stream>>>(cw, cb, Bswz);
    anfis_gemm<<<192, 128, 0, stream>>>(Amem, Cswz, (const v16h*)Bswz, numer);
    anfis_final<<<1, 256, 0, stream>>>(x, Sa, Sc, numer, out);
}
